// CachedAttention_59081570123849
// MI455X (gfx1250) — matrix operand order repaired and checked by reading, not compiled, not run on hardware
//
#include <hip/hip_runtime.h>
#include <hip/hip_bf16.h>

// ---------------------------------------------------------------------------
// CachedAttention for MI455X (gfx1250), wave32 + WMMA bf16 + TDM async loads.
//   B=2, S=2048, D=2048, H=16, Dh=128
// Pipeline:
//   1) fp32 -> bf16 casts (compute-bound problem; bf16 WMMA is the fast path)
//   2) WMMA GEMM (TDM double-buffered LDS tiles): q/k/v projections
//   3) flash attention (TDM K tiles, WMMA QK^T + online softmax + WMMA P*V)
//   4) WMMA GEMM: output projection (+bias) -> fp32 d_out
// ---------------------------------------------------------------------------

typedef __attribute__((ext_vector_type(16))) __bf16 v16bf;
typedef __attribute__((ext_vector_type(8)))  __bf16 v8bf;
typedef __attribute__((ext_vector_type(8)))  float  v8f;
typedef __attribute__((ext_vector_type(4)))  unsigned int v4u;
typedef __attribute__((ext_vector_type(8)))  int    v8i;
typedef __attribute__((ext_vector_type(4)))  int    v4i;

#define D_MODEL   2048
#define NUM_HEADS 16
#define HEAD_DIM  128
#define BATCH     2
#define SEQ       2048
#define MROWS     (BATCH * SEQ)

static __device__ __forceinline__ v16bf cat8(v8bf lo, v8bf hi) {
  return __builtin_shufflevector(lo, hi, 0,1,2,3,4,5,6,7,8,9,10,11,12,13,14,15);
}
static __device__ __forceinline__ v8f wmma_bf16(v16bf a, v16bf b, v8f c) {
  // D = A(16x32 bf16) * B(32x16 bf16) + C(16x16 f32)
  return __builtin_amdgcn_wmma_f32_16x16x32_bf16(false, a, false, b,
                                                 (short)0, c, false, false);
}

// --------------------------- TDM 2-D tile load ------------------------------
// D# per CDNA5 ISA. 2-byte elements, 2-D tile,
// hardware LDS row padding to keep the bank-friendly padded row strides.
//   padIntervalCode: pad after 8*2^code bytes   (3 -> 64B row, 5 -> 256B row)
//   padAmountCode:   pad (code+1) DWORDs        (3 -> +16B)
static __device__ __forceinline__ void tdm_load_2d(
    const void* gsrc, unsigned ldsOff, unsigned tile0, unsigned tile1,
    unsigned tensorDim0, unsigned long long rowStride,
    int padIntervalCode, int padAmountCode) {
  unsigned long long ga = (unsigned long long)gsrc;
  v4u g0 = { 1u,                                   // count=1, user mode
             ldsOff,                               // lds_addr (bytes)
             (unsigned)ga,                         // global_addr[31:0]
             (unsigned)((ga >> 32) & 0x01FFFFFFu)  // global_addr[56:32]
                 | (2u << 30) };                   // type=2 ("image")
  const unsigned td1 = 1u << 28;                   // huge dim1 (no OOB clip)
  unsigned w0 = (1u << 16)                         // data_size = 2 bytes
              | (1u << 20)                         // pad_enable
              | ((unsigned)padIntervalCode << 22)
              | ((unsigned)padAmountCode << 25);
  v8i g1 = { (int)w0,
             (int)((tensorDim0 & 0xFFFFu) << 16),                    // dim0 lo
             (int)(((tensorDim0 >> 16) & 0xFFFFu) |
                   ((td1 & 0xFFFFu) << 16)),                         // dim0 hi | dim1 lo
             (int)(((td1 >> 16) & 0xFFFFu) | (tile0 << 16)),         // dim1 hi | tile0
             (int)(tile1 & 0xFFFFu),                                 // tile1 (tile2=0)
             (int)(unsigned)(rowStride & 0xFFFFFFFFull),             // dim0_stride lo
             (int)(unsigned)((rowStride >> 32) & 0xFFFFull),         // dim0_stride hi
             0 };
  v4i z4 = {0, 0, 0, 0};
  v8i z8 = {0, 0, 0, 0, 0, 0, 0, 0};
  // amdgpu-toolchain (clang-23) 6-arg form
  __builtin_amdgcn_tensor_load_to_lds(g0, g1, z4, z4, z8, 0);
}

// --------------------------- fp32 -> bf16 cast ------------------------------
__global__ void cvt_f32_bf16(const float* __restrict__ in,
                             __bf16* __restrict__ out, int n) {
  int i = blockIdx.x * blockDim.x + threadIdx.x;
  int stride = gridDim.x * blockDim.x;
  for (; i < n; i += stride) out[i] = (__bf16)in[i];
}

// --------------------------- WMMA GEMM --------------------------------------
// C[M,N] = A[M,K] @ W[N,K]^T + bias[N]   (torch Linear semantics)
// Block: 256 threads = 8 waves, block tile 64(M) x 128(N), K-step 32.
// TDM double-buffered LDS tiles; each wave computes 2x2 WMMA tiles (32x32).
template <typename OutT>
__global__ __launch_bounds__(256) void gemm_wmma(
    const __bf16* __restrict__ A, const __bf16* __restrict__ W,
    const float* __restrict__ bias, OutT* __restrict__ C,
    int M, int N, int K) {
  __shared__ __bf16 At[2][64][40];   // 40-el rows via TDM pad (16B aligned)
  __shared__ __bf16 Bt[2][128][40];

  const int tid  = threadIdx.x;
  const int lane = tid & 31;
  const int w    = tid >> 5;
  const int wm   = w >> 2;           // 0..1
  const int wn   = w & 3;            // 0..3
  const int l15  = lane & 15;
  const int g    = lane >> 4;
  const int mBase = blockIdx.y * 64;
  const int nBase = blockIdx.x * 128;

  const unsigned atOff[2] = { (unsigned)(size_t)&At[0][0][0],
                              (unsigned)(size_t)&At[1][0][0] };
  const unsigned btOff[2] = { (unsigned)(size_t)&Bt[0][0][0],
                              (unsigned)(size_t)&Bt[1][0][0] };

  auto issue = [&](int k0, int b) {
    tdm_load_2d(&A[(size_t)mBase * K + k0], atOff[b], 32, 64,
                (unsigned)K, (unsigned long long)K, 3, 3);
    tdm_load_2d(&W[(size_t)nBase * K + k0], btOff[b], 32, 128,
                (unsigned)K, (unsigned long long)K, 3, 3);
  };

  if (w == 0) issue(0, 0);

  v8f acc[2][2] = {};
  int buf = 0;
  for (int k0 = 0; k0 < K; k0 += 32) {
    if (w == 0) {
      if (k0 + 32 < K) {
        issue(k0 + 32, buf ^ 1);                    // prefetch next tile pair
        __builtin_amdgcn_s_wait_tensorcnt(2);       // current pair complete
      } else {
        __builtin_amdgcn_s_wait_tensorcnt(0);
      }
    }
    __syncthreads();

    v16bf af[2], bfr[2];
#pragma unroll
    for (int mt = 0; mt < 2; ++mt) {   // A frag: lane=row, K split at +0/+16
      int r = wm * 32 + mt * 16 + l15;
      af[mt] = cat8(*(const v8bf*)&At[buf][r][g * 8],
                    *(const v8bf*)&At[buf][r][16 + g * 8]);
    }
#pragma unroll
    for (int nt = 0; nt < 2; ++nt) {   // B frag: lane=col, K split at +0/+16
      int r = wn * 32 + nt * 16 + l15;
      bfr[nt] = cat8(*(const v8bf*)&Bt[buf][r][g * 8],
                     *(const v8bf*)&Bt[buf][r][16 + g * 8]);
    }
#pragma unroll
    for (int mt = 0; mt < 2; ++mt)
#pragma unroll
      for (int nt = 0; nt < 2; ++nt)
        acc[mt][nt] = wmma_bf16(af[mt], bfr[nt], acc[mt][nt]);
    __syncthreads();                   // readers done before buffer rewrite
    buf ^= 1;
  }

  // epilogue: C/D layout row = v + 8*(lane>=16), col = lane&15
#pragma unroll
  for (int mt = 0; mt < 2; ++mt)
#pragma unroll
    for (int nt = 0; nt < 2; ++nt)
#pragma unroll
      for (int v = 0; v < 8; ++v) {
        int row = mBase + wm * 32 + mt * 16 + v + 8 * g;
        int col = nBase + wn * 32 + nt * 16 + l15;
        float val = acc[mt][nt][v] + bias[col];
        C[(size_t)row * N + col] = (OutT)val;
      }
}

// --------------------------- flash attention --------------------------------
// Block = 256 threads / 8 waves. Each block: one (b,h), 128 query rows
// (16 per wave). Key loop in chunks of 32: K chunk via TDM (overlapped with
// manual transposed-V staging), WMMA scores, online softmax, WMMA P*V.
__global__ __launch_bounds__(256) void attn_flash(
    const __bf16* __restrict__ qB, const __bf16* __restrict__ kB,
    const __bf16* __restrict__ vB, __bf16* __restrict__ ctx) {
  __shared__ __bf16 Kt[32][136];      // 136-el rows via TDM pad (16B mult)
  __shared__ __bf16 Vt[HEAD_DIM][40]; // V chunk transposed: [d][key]
  __shared__ __bf16 Pt[8][16 * 32];   // per-wave probs scratch (C->A reshape)

  const int tid  = threadIdx.x;
  const int lane = tid & 31;
  const int w    = tid >> 5;
  const int l15  = lane & 15;
  const int g    = lane >> 4;
  const int qtile = blockIdx.x, h = blockIdx.y, b = blockIdx.z;
  const int qbase = qtile * 128;
  const size_t rowOff = (size_t)b * SEQ;
  const int hOff = h * HEAD_DIM;
  const float rs = 0.08838834764831845f;  // 1/sqrt(128)
  const unsigned ktOff = (unsigned)(size_t)&Kt[0][0];

  // Q fragments for this wave's 16 rows (4 K-chunks over d=128), resident
  v16bf qa[4];
  {
    size_t r = rowOff + qbase + w * 16 + l15;
    const __bf16* qp = &qB[r * D_MODEL + hOff];
#pragma unroll
    for (int dc = 0; dc < 4; ++dc)
      qa[dc] = cat8(*(const v8bf*)&qp[dc * 32 + g * 8],
                    *(const v8bf*)&qp[dc * 32 + 16 + g * 8]);
  }

  v8f O[8] = {};
  float m_r[8], l_r[8];
#pragma unroll
  for (int v = 0; v < 8; ++v) { m_r[v] = -INFINITY; l_r[v] = 0.f; }
  __bf16* Pw = &Pt[w][0];

  for (int k0 = 0; k0 < SEQ; k0 += 32) {
    // async K chunk 32x128 -> LDS (row pad 256B->272B), overlapped with V
    if (w == 0)
      tdm_load_2d(&kB[(rowOff + k0) * D_MODEL + hOff], ktOff, HEAD_DIM, 32,
                  (unsigned)D_MODEL, (unsigned long long)D_MODEL, 5, 3);

    // stage V chunk transposed: Vt[d][key]
#pragma unroll
    for (int j = 0; j < 2; ++j) {
      int c = tid + j * 256, r = c >> 4, cc = (c & 15) * 8;
      v8bf vv = *(const v8bf*)&vB[(rowOff + k0 + r) * D_MODEL + hOff + cc];
      if (k0 + 32 < SEQ)
        __builtin_prefetch(&vB[(rowOff + k0 + 32 + r) * D_MODEL + hOff + cc],
                           0, 1);
#pragma unroll
      for (int i = 0; i < 8; ++i) Vt[cc + i][r] = vv[i];
    }
    if (w == 0) __builtin_amdgcn_s_wait_tensorcnt(0);
    __syncthreads();

    // scores: two 16x16 tiles over the 32 keys, contraction over d=128
    v8f s[2] = {};
#pragma unroll
    for (int kt = 0; kt < 2; ++kt) {
      int key = kt * 16 + l15;
#pragma unroll
      for (int dc = 0; dc < 4; ++dc) {
        v16bf kb = cat8(*(const v8bf*)&Kt[key][dc * 32 + g * 8],
                        *(const v8bf*)&Kt[key][dc * 32 + 16 + g * 8]);
        s[kt] = wmma_bf16(qa[dc], kb, s[kt]);
      }
    }

    // online softmax; row r = v + 8*g lives across the 16 lanes of each half
#pragma unroll
    for (int v = 0; v < 8; ++v) {
      float a0 = s[0][v] * rs, a1 = s[1][v] * rs;
      float mx = fmaxf(a0, a1);
      mx = fmaxf(mx, __shfl_xor(mx, 1, 16));
      mx = fmaxf(mx, __shfl_xor(mx, 2, 16));
      mx = fmaxf(mx, __shfl_xor(mx, 4, 16));
      mx = fmaxf(mx, __shfl_xor(mx, 8, 16));
      float mn = fmaxf(m_r[v], mx);
      float alpha = __expf(m_r[v] - mn);
      m_r[v] = mn;
      a0 = __expf(a0 - mn);
      a1 = __expf(a1 - mn);
      float sum = a0 + a1;
      sum += __shfl_xor(sum, 1, 16);
      sum += __shfl_xor(sum, 2, 16);
      sum += __shfl_xor(sum, 4, 16);
      sum += __shfl_xor(sum, 8, 16);
      l_r[v] = l_r[v] * alpha + sum;
#pragma unroll
      for (int nt = 0; nt < 8; ++nt) O[nt][v] *= alpha;
      int prow = v + 8 * g;                       // C-layout -> LDS
      Pw[prow * 32 + l15]      = (__bf16)a0;
      Pw[prow * 32 + 16 + l15] = (__bf16)a1;
    }
    asm volatile("s_wait_dscnt 0x0" ::: "memory"); // wave-private LDS RAW

    // reload P in A-fragment layout (16x32)
    v16bf pa = cat8(*(const v8bf*)&Pw[l15 * 32 + g * 8],
                    *(const v8bf*)&Pw[l15 * 32 + 16 + g * 8]);

    // O += P @ V   (contraction over 32 keys, 8 d-tiles of 16)
#pragma unroll
    for (int nt = 0; nt < 8; ++nt) {
      int dcol = nt * 16 + l15;
      v16bf vbf = cat8(*(const v8bf*)&Vt[dcol][g * 8],
                       *(const v8bf*)&Vt[dcol][16 + g * 8]);
      O[nt] = wmma_bf16(pa, vbf, O[nt]);
    }
    __syncthreads();
  }

  // normalize + write context (bf16) back in [B,S,D] layout
#pragma unroll
  for (int v = 0; v < 8; ++v) {
    float inv = 1.0f / l_r[v];
    size_t row = rowOff + qbase + w * 16 + v + 8 * g;
#pragma unroll
    for (int nt = 0; nt < 8; ++nt) {
      int col = hOff + nt * 16 + l15;
      ctx[row * D_MODEL + col] = (__bf16)(O[nt][v] * inv);
    }
  }
}

// --------------------------- host launcher ----------------------------------
extern "C" void kernel_launch(void* const* d_in, const int* in_sizes, int n_in,
                              void* d_out, int out_size, void* d_ws,
                              size_t ws_size, hipStream_t stream) {
  const float* query  = (const float*)d_in[0];
  const float* key_in = (const float*)d_in[1];
  const float* value  = (const float*)d_in[2];
  const float* Wq = (const float*)d_in[3];
  const float* bq = (const float*)d_in[4];
  const float* Wk = (const float*)d_in[5];
  const float* bk = (const float*)d_in[6];
  const float* Wv = (const float*)d_in[7];
  const float* bv = (const float*)d_in[8];
  const float* Wo = (const float*)d_in[9];
  const float* bo = (const float*)d_in[10];
  float* out = (float*)d_out;

  // workspace carve-out (~151 MB total)
  char* ws = (char*)d_ws;
  size_t off = 0;
  auto alloc = [&](size_t bytes) -> char* {
    char* p = ws + off;
    off += (bytes + 255) & ~(size_t)255;
    return p;
  };
  const size_t actB = (size_t)MROWS * D_MODEL * sizeof(__bf16);
  const size_t wB   = (size_t)D_MODEL * D_MODEL * sizeof(__bf16);
  __bf16* x16[3];
  for (int i = 0; i < 3; ++i) x16[i] = (__bf16*)alloc(actB);
  __bf16* w16[4];
  for (int i = 0; i < 4; ++i) w16[i] = (__bf16*)alloc(wB);
  __bf16* qBuf = (__bf16*)alloc(actB);
  __bf16* kBuf = (__bf16*)alloc(actB);
  __bf16* vBuf = (__bf16*)alloc(actB);
  __bf16* cBuf = (__bf16*)alloc(actB);

  const int nAct = MROWS * D_MODEL;
  const int nW   = D_MODEL * D_MODEL;
  cvt_f32_bf16<<<2048, 256, 0, stream>>>(query,  x16[0], nAct);
  cvt_f32_bf16<<<2048, 256, 0, stream>>>(key_in, x16[1], nAct);
  cvt_f32_bf16<<<2048, 256, 0, stream>>>(value,  x16[2], nAct);
  cvt_f32_bf16<<<2048, 256, 0, stream>>>(Wq, w16[0], nW);
  cvt_f32_bf16<<<2048, 256, 0, stream>>>(Wk, w16[1], nW);
  cvt_f32_bf16<<<2048, 256, 0, stream>>>(Wv, w16[2], nW);
  cvt_f32_bf16<<<2048, 256, 0, stream>>>(Wo, w16[3], nW);

  dim3 gGemm(D_MODEL / 128, MROWS / 64);  // (16, 64)
  gemm_wmma<__bf16><<<gGemm, 256, 0, stream>>>(x16[0], w16[0], bq, qBuf,
                                               MROWS, D_MODEL, D_MODEL);
  gemm_wmma<__bf16><<<gGemm, 256, 0, stream>>>(x16[1], w16[1], bk, kBuf,
                                               MROWS, D_MODEL, D_MODEL);
  gemm_wmma<__bf16><<<gGemm, 256, 0, stream>>>(x16[2], w16[2], bv, vBuf,
                                               MROWS, D_MODEL, D_MODEL);

  dim3 gAttn(SEQ / 128, NUM_HEADS, BATCH);
  attn_flash<<<gAttn, 256, 0, stream>>>(qBuf, kBuf, vBuf, cBuf);

  gemm_wmma<float><<<gGemm, 256, 0, stream>>>(cBuf, w16[3], bo, out,
                                              MROWS, D_MODEL, D_MODEL);
}
